// IDXST_IDCT_80599356277183
// MI455X (gfx1250) — matrix operand order repaired and checked by reading, not compiled, not run on hardware
//
#include <hip/hip_runtime.h>

// ---------------------------------------------------------------------------
// IDXST (dim 0) + IDCT (dim 1) of a 4096x4096 fp32 matrix, expressed as
//   OUT = S * X * C,   S[u,k] = sin(pi*k*(2u+1)/(2M)),  C[j,v] = cos(pi*j*(2v+1)/(2N))
// Two dense GEMMs on the CDNA5 WMMA pipes (v_wmma_f32_16x16x32_bf16) using a
// hi/lo bf16 split (3 significant products) for ~fp32 accuracy with fp32
// accumulation. Twiddle operands are generated in-register from exact integer
// phase reduction + complex-rotation chains (no twiddle memory traffic).
// The global->LDS staging is software-pipelined: step k+1's tile is prefetched
// into VGPRs while step k's WMMAs run, hiding HBM/L2 latency behind the XDL.
// ---------------------------------------------------------------------------

typedef __attribute__((ext_vector_type(16))) __bf16 v16bf;
typedef __attribute__((ext_vector_type(8)))  __bf16 v8bf;
typedef __attribute__((ext_vector_type(2)))  __bf16 v2bf;
typedef __attribute__((ext_vector_type(8)))  float  v8f;

#define PI_F 3.14159265358979323846f
// LDS row pitch in bf16 elements: 80 bytes -> rows stay 16B aligned for
// ds_read_b128 fragment loads, and the pitch breaks bank-conflict patterns.
#define LROW 40

__device__ __forceinline__ void bsplit(float v, __bf16& hi, __bf16& lo) {
  hi = (__bf16)v;                 // RNE to bf16 (8 mantissa bits)
  lo = (__bf16)(v - (float)hi);   // residual: next ~8 bits
}

__device__ __forceinline__ v16bf fragCat(const __bf16* r0, const __bf16* r1) {
  v8bf a = *(const v8bf*)r0;      // 16B aligned -> ds_load_b128
  v8bf b = *(const v8bf*)r1;
  v16bf r;
#pragma unroll
  for (int i = 0; i < 8; ++i) { r[i] = a[i]; r[i + 8] = b[i]; }
  return r;
}

__device__ __forceinline__ v8f wmma_bf16(v16bf a, v16bf b, v8f c) {
  // D = A(16x32 bf16) x B(32x16 bf16) + C(16x16 f32)
  return __builtin_amdgcn_wmma_f32_16x16x32_bf16(false, a, false, b,
                                                 (short)0, c, false, false);
}

// ---------------------------------------------------------------------------
// Kernel 1:  Y1[u,n] = sum_k sin(pi*k*(2u+1)/(2M)) * X[k,n]
//   A operand: generated sin twiddle (per-lane rotation chain)
//   B operand: X tile staged to LDS, transposed + hi/lo-split at staging time
// Block tile 128(M) x 128(N); 8 waves; wave tile = 2 Mtiles x 4 Ntiles.
// ---------------------------------------------------------------------------
__global__ __launch_bounds__(256) void idxst_rows_kernel(
    const float* __restrict__ x, float* __restrict__ y1, int M, int Ncols) {
  __shared__ __align__(16) __bf16 sHi[128 * LROW];
  __shared__ __align__(16) __bf16 sLo[128 * LROW];

  const int t    = threadIdx.x;
  const int lane = t & 31;
  const int wv   = t >> 5;      // 0..7
  const int wm   = wv & 3;      // 4 wave rows * 32 M each
  const int wn   = wv >> 2;     // 2 wave cols * 64 N each
  const int mLn  = lane & 15;   // A/B/C lane index (M or N)
  const int hf   = lane >> 4;   // lane half selects K sub-block

  const int rowBlk = blockIdx.y * 128;
  const int colBlk = blockIdx.x * 128;
  const int rBase  = rowBlk + wm * 32;
  const int cBase  = colBlk + wn * 64;

  // Rotation constants per Mtile: cis(phi) and cis(9*phi), phi = pi*(2u+1)/(2M).
  // A-layout K sequence per lane: 8h+0..7, then jump +9 to 16+8h, then +1 x7.
  float c1[2], s1[2], c9[2], s9[2];
  int   twoU[2];
#pragma unroll
  for (int mt = 0; mt < 2; ++mt) {
    int u  = rBase + mt * 16 + mLn;
    int tu = 2 * u + 1;
    twoU[mt] = tu;
    float phi = PI_F * (float)tu / (2.0f * (float)M);
    sincosf(phi, &s1[mt], &c1[mt]);
    long long p9 = ((long long)9 * (long long)tu) % (long long)(4 * M);
    float th9 = (float)p9 * (PI_F / (2.0f * (float)M));
    sincosf(th9, &s9[mt], &c9[mt]);
  }

  v8f acc[2][4];
#pragma unroll
  for (int mt = 0; mt < 2; ++mt)
#pragma unroll
    for (int nt = 0; nt < 4; ++nt) acc[mt][nt] = v8f{};

  // ---- software-pipelined staging: prefetch tile k0 into registers --------
  float pf[16];
#pragma unroll
  for (int p = 0; p < 8; ++p) {
    int q  = t + 256 * p;   // 0..2047
    int n  = q & 127;
    int k2 = q >> 7;        // 0..15 (pairs of k)
    const float* gp = x + (size_t)(2 * k2) * Ncols + (colBlk + n);
    pf[2 * p]     = gp[0];
    pf[2 * p + 1] = gp[Ncols];
  }

  for (int k0 = 0; k0 < M; k0 += 32) {
    // ---- split the prefetched tile and write LDS as [n][k], hi/lo bf16
#pragma unroll
    for (int p = 0; p < 8; ++p) {
      int q  = t + 256 * p;
      int n  = q & 127;
      int k2 = q >> 7;
      __bf16 h0, l0, h1, l1;
      bsplit(pf[2 * p], h0, l0);
      bsplit(pf[2 * p + 1], h1, l1);
      v2bf ph = {h0, h1}, pl = {l0, l1};
      *(v2bf*)&sHi[n * LROW + 2 * k2] = ph;
      *(v2bf*)&sLo[n * LROW + 2 * k2] = pl;
    }
    __syncthreads();

    // ---- issue next tile's global loads now; they complete under the WMMAs
    if (k0 + 32 < M) {
#pragma unroll
      for (int p = 0; p < 8; ++p) {
        int q  = t + 256 * p;
        int n  = q & 127;
        int k2 = q >> 7;
        const float* gp = x + (size_t)(k0 + 32 + 2 * k2) * Ncols + (colBlk + n);
        pf[2 * p]     = gp[0];
        pf[2 * p + 1] = gp[Ncols];
      }
    }

    // ---- generate A fragments: sin(phi_u * k), exact integer phase seed
    v16bf aHi[2], aLo[2];
#pragma unroll
    for (int mt = 0; mt < 2; ++mt) {
      long long ph0 = ((long long)twoU[mt] * (long long)(k0 + 8 * hf)) %
                      (long long)(4 * M);
      float th = (float)ph0 * (PI_F / (2.0f * (float)M));
      float c, s;
      sincosf(th, &s, &c);
#pragma unroll
      for (int e = 0; e < 16; ++e) {
        __bf16 h, l;
        bsplit(s, h, l);              // value is sin
        aHi[mt][e] = h; aLo[mt][e] = l;
        float C = (e == 7) ? c9[mt] : c1[mt];   // K jump of 9 after e==7
        float S = (e == 7) ? s9[mt] : s1[mt];
        float nc = c * C - s * S;
        float ns = s * C + c * S;
        c = nc; s = ns;
      }
    }

    // ---- B fragments from LDS; 3 significant products per (mt,nt)
#pragma unroll
    for (int nt = 0; nt < 4; ++nt) {
      const int nrow = wn * 64 + nt * 16 + mLn;     // local column of X tile
      const __bf16* bh = &sHi[nrow * LROW + 8 * hf];
      const __bf16* bl = &sLo[nrow * LROW + 8 * hf];
      v16bf bHi = fragCat(bh, bh + 16);
      v16bf bLo = fragCat(bl, bl + 16);
#pragma unroll
      for (int mt = 0; mt < 2; ++mt) {
        acc[mt][nt] = wmma_bf16(aHi[mt], bHi, acc[mt][nt]);
        acc[mt][nt] = wmma_bf16(aHi[mt], bLo, acc[mt][nt]);
        acc[mt][nt] = wmma_bf16(aLo[mt], bHi, acc[mt][nt]);
      }
    }
    __syncthreads();
  }

  // ---- store: C/D layout M = r + 8*hf, N = mLn
#pragma unroll
  for (int mt = 0; mt < 2; ++mt)
#pragma unroll
    for (int nt = 0; nt < 4; ++nt)
#pragma unroll
      for (int r = 0; r < 8; ++r) {
        int row = rBase + mt * 16 + 8 * hf + r;
        int col = cBase + nt * 16 + mLn;
        y1[(size_t)row * Ncols + col] = acc[mt][nt][r];
      }
}

// ---------------------------------------------------------------------------
// Kernel 2:  OUT[u,v] = sum_j Y1[u,j] * cos(pi*j*(2v+1)/(2N))
//   A operand: Y1 tile staged to LDS (row-major [m][k]), hi/lo split
//   B operand: generated cos twiddle
// Wave tile = 4 Mtiles x 2 Ntiles (generation on the 2-tile side).
// ---------------------------------------------------------------------------
__global__ __launch_bounds__(256) void idct_cols_kernel(
    const float* __restrict__ y1, float* __restrict__ out, int Mrows, int Ncols) {
  (void)Mrows;
  __shared__ __align__(16) __bf16 sHi[128 * LROW];
  __shared__ __align__(16) __bf16 sLo[128 * LROW];

  const int t    = threadIdx.x;
  const int lane = t & 31;
  const int wv   = t >> 5;
  const int wm   = wv & 1;      // 2 wave rows * 64 M each
  const int wn   = wv >> 1;     // 4 wave cols * 32 N each
  const int mLn  = lane & 15;
  const int hf   = lane >> 4;

  const int rowBlk = blockIdx.y * 128;
  const int colBlk = blockIdx.x * 128;
  const int rBase  = rowBlk + wm * 64;
  const int cBase  = colBlk + wn * 32;

  // B-layout K sequence per lane is contiguous: 16*hf + e -> single cis(phi) chain.
  float c1[2], s1[2];
  int   twoV[2];
#pragma unroll
  for (int nt = 0; nt < 2; ++nt) {
    int v  = cBase + nt * 16 + mLn;
    int tv = 2 * v + 1;
    twoV[nt] = tv;
    float phi = PI_F * (float)tv / (2.0f * (float)Ncols);
    sincosf(phi, &s1[nt], &c1[nt]);
  }

  v8f acc[4][2];
#pragma unroll
  for (int mt = 0; mt < 4; ++mt)
#pragma unroll
    for (int nt = 0; nt < 2; ++nt) acc[mt][nt] = v8f{};

  // ---- software-pipelined staging: prefetch tile k0=0 into registers ------
  float2 pf[8];
#pragma unroll
  for (int p = 0; p < 8; ++p) {
    int q  = t + 256 * p;
    int k2 = q & 15;
    int m  = q >> 4;          // 0..127
    pf[p] = *(const float2*)(y1 + (size_t)(rowBlk + m) * Ncols + (2 * k2));
  }

  for (int k0 = 0; k0 < Ncols; k0 += 32) {
    // ---- split the prefetched tile and write LDS as [m][k], hi/lo bf16
#pragma unroll
    for (int p = 0; p < 8; ++p) {
      int q  = t + 256 * p;
      int k2 = q & 15;
      int m  = q >> 4;
      __bf16 h0, l0, h1, l1;
      bsplit(pf[p].x, h0, l0);
      bsplit(pf[p].y, h1, l1);
      v2bf ph = {h0, h1}, pl = {l0, l1};
      *(v2bf*)&sHi[m * LROW + 2 * k2] = ph;
      *(v2bf*)&sLo[m * LROW + 2 * k2] = pl;
    }
    __syncthreads();

    // ---- issue next tile's global loads; they complete under the WMMAs
    if (k0 + 32 < Ncols) {
#pragma unroll
      for (int p = 0; p < 8; ++p) {
        int q  = t + 256 * p;
        int k2 = q & 15;
        int m  = q >> 4;
        pf[p] = *(const float2*)(y1 + (size_t)(rowBlk + m) * Ncols +
                                 (k0 + 32 + 2 * k2));
      }
    }

    // ---- generate B fragments: cos(phi_v * j)
    v16bf bHi[2], bLo[2];
#pragma unroll
    for (int nt = 0; nt < 2; ++nt) {
      long long ph0 = ((long long)twoV[nt] * (long long)(k0 + 16 * hf)) %
                      (long long)(4 * Ncols);
      float th = (float)ph0 * (PI_F / (2.0f * (float)Ncols));
      float c, s;
      sincosf(th, &s, &c);
#pragma unroll
      for (int e = 0; e < 16; ++e) {
        __bf16 h, l;
        bsplit(c, h, l);              // value is cos
        bHi[nt][e] = h; bLo[nt][e] = l;
        float nc = c * c1[nt] - s * s1[nt];
        float ns = s * c1[nt] + c * s1[nt];
        c = nc; s = ns;
      }
    }

    // ---- A fragments from LDS: contiguous run at k = 16*hf+0..15
#pragma unroll
    for (int mt = 0; mt < 4; ++mt) {
      const int mrow = wm * 64 + mt * 16 + mLn;
      const __bf16* ah = &sHi[mrow * LROW + 16 * hf];
      const __bf16* al = &sLo[mrow * LROW + 16 * hf];
      v16bf aHi = fragCat(ah, ah + 8);
      v16bf aLo = fragCat(al, al + 8);
#pragma unroll
      for (int nt = 0; nt < 2; ++nt) {
        acc[mt][nt] = wmma_bf16(aHi, bHi[nt], acc[mt][nt]);
        acc[mt][nt] = wmma_bf16(aHi, bLo[nt], acc[mt][nt]);
        acc[mt][nt] = wmma_bf16(aLo, bHi[nt], acc[mt][nt]);
      }
    }
    __syncthreads();
  }

#pragma unroll
  for (int mt = 0; mt < 4; ++mt)
#pragma unroll
    for (int nt = 0; nt < 2; ++nt)
#pragma unroll
      for (int r = 0; r < 8; ++r) {
        int row = rBase + mt * 16 + 8 * hf + r;
        int col = cBase + nt * 16 + mLn;
        out[(size_t)row * Ncols + col] = acc[mt][nt][r];
      }
}

extern "C" void kernel_launch(void* const* d_in, const int* in_sizes, int n_in,
                              void* d_out, int out_size, void* d_ws, size_t ws_size,
                              hipStream_t stream) {
  (void)n_in; (void)out_size; (void)ws_size;
  const float* x = (const float*)d_in[0];
  // expkM/expkN (d_in[1], d_in[2]) are regenerated on the fly; only dims used.
  const int M = in_sizes[1] / 2;   // 4096
  const int N = in_sizes[2] / 2;   // 4096

  float* y1  = (float*)d_ws;       // M*N fp32 intermediate (64 MB for 4096^2)
  float* out = (float*)d_out;

  dim3 blk(256);
  dim3 grd(N / 128, M / 128);
  idxst_rows_kernel<<<grd, blk, 0, stream>>>(x, y1, M, N);
  idct_cols_kernel<<<grd, blk, 0, stream>>>(y1, out, M, N);
}
